// CabbageEnhancedOneFormer3D_60876866453847
// MI455X (gfx1250) — matrix operand order repaired and checked by reading, not compiled, not run on hardware
//
#include <hip/hip_runtime.h>
#include <math.h>

typedef __attribute__((ext_vector_type(16))) _Float16 v16h;
typedef __attribute__((ext_vector_type(8)))  float    v8f;
typedef __attribute__((ext_vector_type(2)))  float    v2f;

// ---------------- cross-lane helpers (wave32) ----------------
// Exchange value with the mirror lane in the other 16-lane half (xor 16),
// as a VALU v_permlanex16 (no LDS pipe) when available.
__device__ __forceinline__ float xor16(float v) {
#if __has_builtin(__builtin_amdgcn_permlanex16)
  int i = __builtin_bit_cast(int, v);
  i = __builtin_amdgcn_permlanex16(i, i, 0x76543210, 0xfedcba98, false, false);
  return __builtin_bit_cast(float, i);
#else
  return __shfl_xor(v, 16, 32);
#endif
}
__device__ __forceinline__ float red32_sum(float v) {
#pragma unroll
  for (int m = 1; m <= 16; m <<= 1) v += __shfl_xor(v, m, 32);
  return v;
}

// ---------------- generic GEMM: Y = act(X @ W^T + bias) ----------------
// X:[M,K] f32 row-major, W:[N,K] f32 row-major, Y:[M,N]
// Wave tile: 16(M) x 64(N): one A fragment feeds 4 WMMAs (v_wmma_f32_16x16x32_f16).
// Block = 128 threads (4 waves = 4 consecutive m-tiles).
// Requires K%32==0, N%64==0, M%64==0.
template<int ACT>  // 0 = none, 1 = relu
__global__ void gemm_wmma_f16(const float* __restrict__ X, const float* __restrict__ W,
                              const float* __restrict__ bias, float* __restrict__ Y,
                              int M, int N, int K) {
  const int lane = threadIdx.x & 31;
  const int wv   = threadIdx.x >> 5;
  const int g    = lane >> 4;      // lane half
  const int li   = lane & 15;
  const int tm   = blockIdx.x * 4 + wv;
  const int tn0  = blockIdx.y * 4;
  const float* Xrow = X + (size_t)(tm * 16 + li) * K;
  const float* Wr[4];
#pragma unroll
  for (int j = 0; j < 4; ++j) Wr[j] = W + (size_t)((tn0 + j) * 16 + li) * K;

  v8f acc[4] = {};
  for (int k0 = 0; k0 < K; k0 += 32) {
    v16h a;
#pragma unroll
    for (int vi = 0; vi < 8; ++vi) {
      // A 16x32 f16 layout: VGPR v<4: K=2v+8g(+1); v>=4: K=16+2(v-4)+8g(+1); M=li
      int ka = k0 + ((vi < 4) ? (2 * vi + 8 * g) : (16 + 2 * (vi - 4) + 8 * g));
      a[2 * vi]     = (_Float16)Xrow[ka];
      a[2 * vi + 1] = (_Float16)Xrow[ka + 1];
    }
#pragma unroll
    for (int j = 0; j < 4; ++j) {
      v16h b;
#pragma unroll
      for (int vi = 0; vi < 8; ++vi) {
        // B 32x16 f16 layout: VGPR v<4: K=2v+8g(+1); v>=4: K=16+2(v-4)+8g(+1); N=li
        int kb = k0 + ((vi < 4) ? (2 * vi + 8 * g) : (16 + 2 * (vi - 4) + 8 * g));
        b[2 * vi]     = (_Float16)Wr[j][kb];
        b[2 * vi + 1] = (_Float16)Wr[j][kb + 1];
      }
      acc[j] = __builtin_amdgcn_wmma_f32_16x16x32_f16(false, a, false, b, (short)0,
                                                      acc[j], false, false);
    }
  }
#pragma unroll
  for (int j = 0; j < 4; ++j) {
    const int n  = (tn0 + j) * 16 + li;
    const float bs = bias[n];
#pragma unroll
    for (int r = 0; r < 8; ++r) {  // C/D: VGPR r -> row M = r + 8g, col N = li
      int m = tm * 16 + r + 8 * g;
      float v = acc[j][r] + bs;
      if (ACT == 1) v = fmaxf(v, 0.f);
      Y[(size_t)m * N + n] = v;
    }
  }
}

// ---------------- flash attention, f32 WMMA 16x16x4, transposed tiles ----------------
// qkv:[B,N,384] (q|k|v concat, head-interleaved), ctx:[B,N,128]
// Computes S^T = K @ Q^T so each softmax row (fixed q) is 8 in-lane values +
// one xor16 exchange; P*V is done as O^T += V^T @ P^T with register-only
// routing of P (8 permlanex16 + selects per tile). No LDS, no barriers.
// Block = 128 (4 independent waves); grid = (N/64, NH, B).
__global__ void flash_attn(const float* __restrict__ qkv, float* __restrict__ ctx,
                           int Npts) {
  const int H3 = 384;
  const int lane = threadIdx.x & 31;
  const int wv   = threadIdx.x >> 5;
  const int g    = lane >> 4;
  const int li   = lane & 15;
  const int bb   = blockIdx.z, hh = blockIdx.y;
  const int qbase = (blockIdx.x * 4 + wv) * 16;
  const float* base = qkv + (size_t)bb * Npts * H3;

  // Q^T as B operand: (K=d, N=q=li): chunk c -> d = 4c+2g (+1); pre-scale 1/sqrt(16)
  v2f qb[4];
  {
    const float* qrow = base + (size_t)(qbase + li) * H3 + hh * 16;
#pragma unroll
    for (int c = 0; c < 4; ++c) {
      qb[c].x = qrow[4 * c + 2 * g] * 0.25f;
      qb[c].y = qrow[4 * c + 2 * g + 1] * 0.25f;
    }
  }
  v8f o = {};                     // O^T: (M=d=r+8g, N=q=li)
  float m = -3.0e38f, l = 0.f;    // per-lane state for q row li (mirrored in halves)

  for (int j = 0; j < Npts; j += 16) {
    // K tile as A operand: (M=key=j+li, K=d)
    v2f ka[4];
    const float* krow = base + (size_t)(j + li) * H3 + 128 + hh * 16;
#pragma unroll
    for (int c = 0; c < 4; ++c) {
      ka[c].x = krow[4 * c + 2 * g];
      ka[c].y = krow[4 * c + 2 * g + 1];
    }
    v8f s = {};
#pragma unroll
    for (int c = 0; c < 4; ++c)
      s = __builtin_amdgcn_wmma_f32_16x16x4_f32(false, ka[c], false, qb[c], (short)0, s,
                                                false, false);
    // s[r] = S^T[key = j + r + 8g][q = li]
    float tmax = s[0];
#pragma unroll
    for (int r = 1; r < 8; ++r) tmax = fmaxf(tmax, s[r]);
    tmax = fmaxf(tmax, xor16(tmax));
    const float mnew = fmaxf(m, tmax);
    const float corr = __expf(m - mnew);
    float p[8]; float ps = 0.f;
#pragma unroll
    for (int r = 0; r < 8; ++r) { p[r] = __expf(s[r] - mnew); ps += p[r]; }
    ps += xor16(ps);
    l = l * corr + ps;
    m = mnew;
#pragma unroll
    for (int r = 0; r < 8; ++r) o[r] *= corr;
    // half-swapped copies of P for B-operand routing
    float px[8];
#pragma unroll
    for (int r = 0; r < 8; ++r) px[r] = xor16(p[r]);
    // O^T += V^T @ P^T : A = V^T (M=d=li, K=key), B = P^T (K=key, N=q=li)
#pragma unroll
    for (int c = 0; c < 4; ++c) {
      v2f va, pb;
      const float* vr = base + (size_t)(j + 4 * c + 2 * g) * H3 + 256 + hh * 16;
      va.x = vr[li];
      va.y = vr[H3 + li];
      if (c < 2) {  // keys 4c,4c+1 in-lane for g=0; keys 4c+2,4c+3 via swap for g=1
        pb.x = g ? px[4 * c + 2] : p[4 * c];
        pb.y = g ? px[4 * c + 3] : p[4 * c + 1];
      } else {      // keys 4c,4c+1 via swap for g=0; keys 4c+2,4c+3 in-lane for g=1
        pb.x = g ? p[4 * c - 6] : px[4 * c - 8];
        pb.y = g ? p[4 * c - 5] : px[4 * c - 7];
      }
      o = __builtin_amdgcn_wmma_f32_16x16x4_f32(false, va, false, pb, (short)0, o,
                                                false, false);
    }
  }
  const float inv_l = 1.f / l;
#pragma unroll
  for (int r = 0; r < 8; ++r)
    ctx[((size_t)(bb * Npts + qbase + li)) * 128 + hh * 16 + r + 8 * g] = o[r] * inv_l;
}

// ---------------- residual add + LayerNorm (H=128), one wave per row ----------------
__global__ void add_ln(float* __restrict__ hbuf, const float* __restrict__ addend,
                       const float* __restrict__ gamma, const float* __restrict__ beta) {
  const int lane = threadIdx.x & 31;
  const int wv   = threadIdx.x >> 5;
  const int row  = blockIdx.x * 8 + wv;
  float* hrow = hbuf + (size_t)row * 128;
  const float* arow = addend + (size_t)row * 128;
  float v[4]; float s = 0.f;
#pragma unroll
  for (int i = 0; i < 4; ++i) { v[i] = hrow[lane + 32 * i] + arow[lane + 32 * i]; s += v[i]; }
  s = red32_sum(s);
  const float mean = s * (1.f / 128.f);
  float t = 0.f;
#pragma unroll
  for (int i = 0; i < 4; ++i) { float d = v[i] - mean; t += d * d; }
  t = red32_sum(t);
  const float rstd = rsqrtf(t * (1.f / 128.f) + 1e-5f);
#pragma unroll
  for (int i = 0; i < 4; ++i) {
    int col = lane + 32 * i;
    hrow[col] = (v[i] - mean) * rstd * gamma[col] + beta[col];
  }
}

// ---------------- tiny heads ----------------
__global__ void head3(const float* __restrict__ X, const float* __restrict__ W,
                      const float* __restrict__ bias, float* __restrict__ out) {
  const int lane = threadIdx.x & 31;
  const int wv   = threadIdx.x >> 5;
  const int row  = blockIdx.x * 8 + wv;
  const float* xr = X + (size_t)row * 128;
  float x[4];
#pragma unroll
  for (int i = 0; i < 4; ++i) x[i] = xr[lane + 32 * i];
#pragma unroll
  for (int c = 0; c < 3; ++c) {
    float s = 0.f;
#pragma unroll
    for (int i = 0; i < 4; ++i) s += x[i] * W[c * 128 + lane + 32 * i];
    s = red32_sum(s);
    if (lane == 0) out[(size_t)row * 3 + c] = s + bias[c];
  }
}

__global__ void det_head(const float* __restrict__ X, const float* __restrict__ W,
                         const float* __restrict__ bias, float* __restrict__ out) {
  const int lane = threadIdx.x & 31;
  const int wv   = threadIdx.x >> 5;
  const int row  = blockIdx.x * 8 + wv;
  const float* xr = X + (size_t)row * 64;
  float s = xr[lane] * W[lane] + xr[lane + 32] * W[lane + 32];
  s = red32_sum(s);
  if (lane == 0) out[row] = 1.f / (1.f + __expf(-(s + bias[0])));
}

__global__ void quality_mean(const float* __restrict__ hp, float* __restrict__ out,
                             int Npts) {
  const int b = blockIdx.x;
  float s = 0.f;
  for (int i = threadIdx.x; i < Npts; i += blockDim.x) s += hp[(size_t)b * Npts + i];
  s = red32_sum(s);
  __shared__ float red[8];
  if ((threadIdx.x & 31) == 0) red[threadIdx.x >> 5] = s;
  __syncthreads();
  if (threadIdx.x < 32) {
    float t = (threadIdx.x < 8) ? red[threadIdx.x] : 0.f;
#pragma unroll
    for (int m = 1; m <= 4; m <<= 1) t += __shfl_xor(t, m, 32);
    if (threadIdx.x == 0) out[b] = t / (float)Npts;
  }
}

// ---------------- host side ----------------
static void launch_gemm(const float* X, const float* W, const float* bias, float* Y,
                        int M, int N, int K, bool relu, hipStream_t stream) {
  dim3 grid(M / 64, N / 64);
  dim3 block(128);
  if (relu) gemm_wmma_f16<1><<<grid, block, 0, stream>>>(X, W, bias, Y, M, N, K);
  else      gemm_wmma_f16<0><<<grid, block, 0, stream>>>(X, W, bias, Y, M, N, K);
}

extern "C" void kernel_launch(void* const* d_in, const int* in_sizes, int n_in,
                              void* d_out, int out_size, void* d_ws, size_t ws_size,
                              hipStream_t stream) {
  (void)in_sizes; (void)n_in; (void)out_size; (void)ws_size;
  const int B = 4, N = 2048, C = 256, H = 128, NH = 8;
  const int M = B * N;

  const float* x = (const float*)d_in[0];
  auto in = [&](int i) { return (const float*)d_in[i]; };

  float* ws      = (float*)d_ws;
  float* buf_h   = ws;                             // [M,H]
  float* buf_qkv = buf_h   + (size_t)M * H;        // [M,3H]
  float* buf_a   = buf_qkv + (size_t)M * 3 * H;    // [M,2H]
  float* buf_b   = buf_a   + (size_t)M * 2 * H;    // [M,H]

  float* out_logits = (float*)d_out;               // [M,3]
  float* out_hp     = out_logits + (size_t)M * 3;  // [M]
  float* out_q      = out_hp + M;                  // [B]

  // feature fusion
  launch_gemm(x,     in(1), in(2), buf_a, M, H, C, true,  stream);
  launch_gemm(buf_a, in(3), in(4), buf_h, M, H, H, false, stream);

  for (int l = 0; l < 2; ++l) {
    const int o = 5 + l * 12;
    // MHA
    launch_gemm(buf_h, in(o + 0), in(o + 1), buf_qkv, M, 3 * H, H, false, stream);
    flash_attn<<<dim3(N / 64, NH, B), 128, 0, stream>>>(buf_qkv, buf_b, N);
    launch_gemm(buf_b, in(o + 2), in(o + 3), buf_a, M, H, H, false, stream);
    add_ln<<<M / 8, 256, 0, stream>>>(buf_h, buf_a, in(o + 8), in(o + 9));
    // FFN
    launch_gemm(buf_h, in(o + 4), in(o + 5), buf_a, M, 2 * H, H, true,  stream);
    launch_gemm(buf_a, in(o + 6), in(o + 7), buf_b, M, H, 2 * H, false, stream);
    add_ln<<<M / 8, 256, 0, stream>>>(buf_h, buf_b, in(o + 10), in(o + 11));
  }

  // output heads
  launch_gemm(buf_h, in(29), in(30), buf_a, M, H, H, true, stream);
  head3<<<M / 8, 256, 0, stream>>>(buf_a, in(31), in(32), out_logits);
  launch_gemm(buf_h, in(33), in(34), buf_b, M, H / 2, H, true, stream);
  det_head<<<M / 8, 256, 0, stream>>>(buf_b, in(35), in(36), out_hp);
  quality_mean<<<B, 256, 0, stream>>>(out_hp, out_q, N);
}
